// Net_25082609009399
// MI455X (gfx1250) — matrix operand order repaired and checked by reading, not compiled, not run on hardware
//
#include <hip/hip_runtime.h>
#include <hip/hip_bf16.h>

// ---------------------------------------------------------------------------
// PointTransformerConv GNN for MI455X (gfx1250, wave32, WMMA).
//  - Dense GEMMs use v_wmma_f32_16x16x32_f16; weights are pre-packed into
//    B-fragment-major layout so every operand load is a contiguous b128.
//  - Each wave computes a 16-row x fo strip: A fragment loaded once per
//    k-block, reused across NT=fo/16 column tiles (NT WMMAs back-to-back).
//  - Edge softmax: two atomic passes, 4 channels/thread (b128 gathers);
//    segment-max via order-preserving uint keys + integer atomicMax.
//  - Node accumulator working set (~150MB at fo=128) ~fits 192MB L2, so the
//    edge passes are L2-atomic bound; GEMMs (~18 GFLOP) are negligible.
// ---------------------------------------------------------------------------

#define NNODES 50000
#define NPAD   50048          // multiple of 64 (rows per GEMM block)
#define NEDGE  800000
#define ETOT   (NEDGE + NNODES)
#define LDA    416            // padded concat width (400 real channels)

typedef __attribute__((ext_vector_type(16))) _Float16 v16h;
typedef __attribute__((ext_vector_type(8)))  _Float16 h8v;
typedef __attribute__((ext_vector_type(8)))  float    v8f;
typedef __attribute__((ext_vector_type(4)))  float    f4v;
typedef __attribute__((ext_vector_type(4)))  unsigned u4v;

// ---- order-preserving float <-> uint key (for integer atomicMax) -----------
__device__ __forceinline__ unsigned fkey(float f) {
    unsigned b = __float_as_uint(f);
    return (b & 0x80000000u) ? ~b : (b | 0x80000000u);
}
__device__ __forceinline__ float fdecode(unsigned k) {
    unsigned b = (k & 0x80000000u) ? (k & 0x7FFFFFFFu) : ~k;
    return __uint_as_float(b);
}
#define NEG_INF_KEY 0x007FFFFFu   // fkey(-inf)

// ---------------------------------------------------------------------------
// WMMA GEMM, fragment-packed weights.
//   A : f16 [NPAD x lda], k = columns 0..32*kblocks-1 (zero padded)
//   Wp: f16 packed fragments: [((nt*kblocks + kb)*32 + lane)*16 + i]
//   C : f32 [NPAD x ldc], ldc = NT*16  (+ optional bias over columns)
// Block = 128 threads = 4 waves; wave w computes rows [(bx*4+w)*16, +16),
// all NT column tiles (A fragment reused across tiles).
// ---------------------------------------------------------------------------
template <int NT>
__global__ __launch_bounds__(128) void wmma_gemm_f16(
    const _Float16* __restrict__ A, int lda,
    const _Float16* __restrict__ Wp,
    float* __restrict__ C, int ldc,
    const float* __restrict__ bias,
    int kblocks)
{
    const int lane = threadIdx.x & 31;
    const int wave = threadIdx.x >> 5;
    const int half = lane >> 4;      // 0: lanes 0-15, 1: lanes 16-31
    const int l15  = lane & 15;
    const int m0   = (blockIdx.x * 4 + wave) * 16;

    v8f acc[NT];
#pragma unroll
    for (int nt = 0; nt < NT; ++nt)
        acc[nt] = (v8f){0.f, 0.f, 0.f, 0.f, 0.f, 0.f, 0.f, 0.f};

    for (int kb = 0; kb < kblocks; ++kb) {
        const int k0 = kb * 32;
        // A fragment (16x32 f16): lanes 0-15 hold K {0..15} of row l15,
        // lanes 16-31 hold K {16..31}. Two contiguous 16B loads.
        const _Float16* ap = A + (size_t)(m0 + l15) * lda + k0 + half * 16;
        h8v alo = *(const h8v*)(ap);
        h8v ahi = *(const h8v*)(ap + 8);
        v16h af;
#pragma unroll
        for (int i = 0; i < 8; ++i) { af[i] = alo[i]; af[8 + i] = ahi[i]; }

#pragma unroll
        for (int nt = 0; nt < NT; ++nt) {
            // Packed B fragment: this lane's 16 halfs are contiguous (32B).
            const v16h bf = *(const v16h*)(
                Wp + ((size_t)(nt * kblocks + kb) * 32 + lane) * 16);
            acc[nt] = __builtin_amdgcn_wmma_f32_16x16x32_f16(
                false, af, false, bf, (short)0, acc[nt], false, false);
        }
    }

    // C/D layout: VGPR j -> row m0 + half*8 + j, col nt*16 + l15.
    const int row0 = m0 + half * 8;
#pragma unroll
    for (int nt = 0; nt < NT; ++nt) {
        const int col = nt * 16 + l15;
        const float bv = bias ? bias[col] : 0.0f;
#pragma unroll
        for (int j = 0; j < 8; ++j)
            C[(size_t)(row0 + j) * ldc + col] = acc[nt][j] + bv;
    }
}

// ---- weight f32 [fi x fo] -> f16 packed B-fragments (zero-padded K) --------
// Packed index: (((nt*kblocks + kb)*32 + lane)*16 + i)
//   col = nt*16 + (lane&15);  k = kb*32 + (lane>>4)*16 + i
__global__ void cvt_w_packed(const float* __restrict__ W,
                             _Float16* __restrict__ Wh,
                             int fi, int fo, int kblocks)
{
    int idx = blockIdx.x * blockDim.x + threadIdx.x;
    int total = (fo / 16) * kblocks * 512;     // == kpad * fo
    if (idx >= total) return;
    int i    = idx & 15;
    int lane = (idx >> 4) & 31;
    int kb   = (idx >> 9) % kblocks;
    int nt   = idx / (kblocks << 9);
    int col  = nt * 16 + (lane & 15);
    int k    = kb * 32 + (lane >> 4) * 16 + i;
    Wh[idx] = (k < fi) ? (_Float16)W[(size_t)k * fo + col] : (_Float16)0.0f;
}

// ---- init padded f16 feature buffer with x (cols 0..15), zeros elsewhere ---
__global__ void init_features(const float* __restrict__ x,
                              _Float16* __restrict__ fh)
{
    int idx = blockIdx.x * blockDim.x + threadIdx.x;
    if (idx >= NPAD * LDA) return;
    int n = idx / LDA, c = idx - n * LDA;
    float v = (n < NNODES && c < 16) ? x[n * 16 + c] : 0.0f;
    fh[idx] = (_Float16)v;
}

// ---- per-layer segment accumulators init -----------------------------------
__global__ void init_seg(unsigned* __restrict__ mkey, float* __restrict__ s,
                         float* __restrict__ num, int total)
{
    int idx = blockIdx.x * blockDim.x + threadIdx.x;
    if (idx >= total) return;
    mkey[idx] = NEG_INF_KEY;
    s[idx]    = 0.0f;
    num[idx]  = 0.0f;
}

// ---- edge pass A: alpha -> segment max; 4 channels per thread --------------
__global__ void edge_max4(const float* __restrict__ x, const int* __restrict__ ei,
                          const float* __restrict__ a_src,
                          const float* __restrict__ a_dst,
                          const float* __restrict__ Wp,
                          const float* __restrict__ bp,
                          unsigned* __restrict__ mkey, int fo)
{
    const int g = fo >> 2;
    int idx = blockIdx.x * blockDim.x + threadIdx.x;
    if (idx >= ETOT * g) return;
    int e = idx / g, c0 = (idx - e * g) << 2;
    int s, d;
    if (e < NEDGE) { s = ei[e]; d = ei[NEDGE + e]; }
    else           { s = d = e - NEDGE; }
    float dx = 0.f, dy = 0.f, dz = 0.f;
    if (s != d) {
        dx = x[d * 16 + 0] - x[s * 16 + 0];
        dy = x[d * 16 + 1] - x[s * 16 + 1];
        dz = x[d * 16 + 2] - x[s * 16 + 2];
    }
    const f4v wx = *(const f4v*)&Wp[c0];
    const f4v wy = *(const f4v*)&Wp[fo + c0];
    const f4v wz = *(const f4v*)&Wp[2 * fo + c0];
    const f4v b4 = *(const f4v*)&bp[c0];
    const f4v ad = *(const f4v*)&a_dst[(size_t)d * fo + c0];
    const f4v as = *(const f4v*)&a_src[(size_t)s * fo + c0];
    unsigned* mp = &mkey[(size_t)d * fo + c0];
#pragma unroll
    for (int j = 0; j < 4; ++j) {
        float delta = dx * wx[j] + dy * wy[j] + dz * wz[j] + b4[j];
        atomicMax(mp + j, fkey(ad[j] - as[j] + delta));
    }
}

// ---- edge pass B: s += e^a ; num += e^a * (v[src] + delta); 4 ch/thread ----
__global__ void edge_acc4(const float* __restrict__ x, const int* __restrict__ ei,
                          const float* __restrict__ a_src,
                          const float* __restrict__ a_dst,
                          const float* __restrict__ vbuf,
                          const float* __restrict__ Wp,
                          const float* __restrict__ bp,
                          const unsigned* __restrict__ mkey,
                          float* __restrict__ sden, float* __restrict__ num,
                          int fo)
{
    const int g = fo >> 2;
    int idx = blockIdx.x * blockDim.x + threadIdx.x;
    if (idx >= ETOT * g) return;
    int e = idx / g, c0 = (idx - e * g) << 2;
    int s, d;
    if (e < NEDGE) { s = ei[e]; d = ei[NEDGE + e]; }
    else           { s = d = e - NEDGE; }
    float dx = 0.f, dy = 0.f, dz = 0.f;
    if (s != d) {
        dx = x[d * 16 + 0] - x[s * 16 + 0];
        dy = x[d * 16 + 1] - x[s * 16 + 1];
        dz = x[d * 16 + 2] - x[s * 16 + 2];
    }
    const f4v wx = *(const f4v*)&Wp[c0];
    const f4v wy = *(const f4v*)&Wp[fo + c0];
    const f4v wz = *(const f4v*)&Wp[2 * fo + c0];
    const f4v b4 = *(const f4v*)&bp[c0];
    const size_t di = (size_t)d * fo + c0;
    const size_t si = (size_t)s * fo + c0;
    const f4v ad = *(const f4v*)&a_dst[di];
    const f4v as = *(const f4v*)&a_src[si];
    const f4v vv = *(const f4v*)&vbuf[si];
    const u4v mk = *(const u4v*)&mkey[di];
#pragma unroll
    for (int j = 0; j < 4; ++j) {
        float delta = dx * wx[j] + dy * wy[j] + dz * wz[j] + b4[j];
        float alpha = ad[j] - as[j] + delta;
        float ea = __expf(alpha - fdecode(mk[j]));
        atomicAdd(&sden[di + j], ea);
        atomicAdd(&num[di + j], ea * (vv[j] + delta));
    }
}

// ---- finalize layer: lrelu(num/s) -> f16 concat slice ----------------------
__global__ void finalize_layer(const float* __restrict__ num,
                               const float* __restrict__ sden,
                               _Float16* __restrict__ fh, int fo, int off)
{
    int idx = blockIdx.x * blockDim.x + threadIdx.x;
    if (idx >= NNODES * fo) return;
    int n = idx / fo, c = idx - n * fo;
    float v = num[idx] / sden[idx];
    v = (v > 0.0f) ? v : 0.01f * v;
    fh[(size_t)n * LDA + off + c] = (_Float16)v;
}

// ---- final tiny MLP layer: hid[N x 64] @ Wm2[64 x 8] + bm2 -----------------
__global__ void mlp_out(const float* __restrict__ hid,
                        const float* __restrict__ Wm2,
                        const float* __restrict__ bm2,
                        float* __restrict__ out)
{
    int n = blockIdx.x * blockDim.x + threadIdx.x;
    if (n >= NNODES) return;
    float o[8];
#pragma unroll
    for (int j = 0; j < 8; ++j) o[j] = bm2[j];
    for (int k = 0; k < 64; ++k) {
        float h = hid[(size_t)n * 64 + k];
#pragma unroll
        for (int j = 0; j < 8; ++j) o[j] += h * Wm2[k * 8 + j];
    }
#pragma unroll
    for (int j = 0; j < 8; ++j) out[(size_t)n * 8 + j] = o[j];
}

// ---------------------------------------------------------------------------
extern "C" void kernel_launch(void* const* d_in, const int* in_sizes, int n_in,
                              void* d_out, int out_size, void* d_ws, size_t ws_size,
                              hipStream_t stream)
{
    (void)in_sizes; (void)n_in; (void)out_size; (void)ws_size;

    const float* x  = (const float*)d_in[0];
    const int*   ei = (const int*)d_in[1];

    // workspace carve-out (256B aligned)
    char* p = (char*)d_ws;
    auto take = [&](size_t nbytes) -> void* {
        void* r = (void*)p;
        p += (nbytes + 255) & ~(size_t)255;
        return r;
    };
    _Float16* fh    = (_Float16*)take((size_t)NPAD * LDA * sizeof(_Float16));
    float*    vbuf  = (float*)   take((size_t)NPAD * 128 * sizeof(float));
    float*    a_src = (float*)   take((size_t)NPAD * 128 * sizeof(float));
    float*    a_dst = (float*)   take((size_t)NPAD * 128 * sizeof(float));
    unsigned* mkey  = (unsigned*)take((size_t)NNODES * 128 * sizeof(unsigned));
    float*    sden  = (float*)   take((size_t)NNODES * 128 * sizeof(float));
    float*    numb  = (float*)   take((size_t)NNODES * 128 * sizeof(float));
    float*    hid   = (float*)   take((size_t)NPAD * 64 * sizeof(float));
    _Float16* wh0   = (_Float16*)take((size_t)416 * 128 * sizeof(_Float16));
    _Float16* wh1   = (_Float16*)take((size_t)416 * 128 * sizeof(_Float16));
    _Float16* wh2   = (_Float16*)take((size_t)416 * 128 * sizeof(_Float16));

    auto launch_gemm = [&](const _Float16* Wpk, float* C, int fo,
                           const float* bias, int kb) {
        dim3 g(NPAD / 64);
        if (fo == 128)
            wmma_gemm_f16<8><<<g, 128, 0, stream>>>(fh, LDA, Wpk, C, fo, bias, kb);
        else
            wmma_gemm_f16<4><<<g, 128, 0, stream>>>(fh, LDA, Wpk, C, fo, bias, kb);
    };

    // features: [x | x1 | x2 | x3 | x4 | zero-pad] as f16, re-zeroed each call
    {
        int tot = NPAD * LDA;
        init_features<<<(tot + 255) / 256, 256, 0, stream>>>(x, fh);
    }

    const int FI[4]   = {16, 80, 144, 272};
    const int FO[4]   = {64, 64, 128, 128};
    const int OFF[4]  = {16, 80, 144, 272};
    const int WIDX[4] = {2, 7, 12, 17};   // Wl,Ws,Wd,Wp,bp per layer

    for (int L = 0; L < 4; ++L) {
        const int fi = FI[L], fo = FO[L];
        const int kpad = ((fi + 31) / 32) * 32;
        const int kb   = kpad / 32;
        const float* pWl = (const float*)d_in[WIDX[L] + 0];
        const float* pWs = (const float*)d_in[WIDX[L] + 1];
        const float* pWd = (const float*)d_in[WIDX[L] + 2];
        const float* pWp = (const float*)d_in[WIDX[L] + 3];
        const float* pbp = (const float*)d_in[WIDX[L] + 4];

        const int wtot = kpad * fo;
        cvt_w_packed<<<(wtot + 255) / 256, 256, 0, stream>>>(pWl, wh0, fi, fo, kb);
        cvt_w_packed<<<(wtot + 255) / 256, 256, 0, stream>>>(pWs, wh1, fi, fo, kb);
        cvt_w_packed<<<(wtot + 255) / 256, 256, 0, stream>>>(pWd, wh2, fi, fo, kb);

        launch_gemm(wh0, vbuf,  fo, nullptr, kb);
        launch_gemm(wh1, a_src, fo, nullptr, kb);
        launch_gemm(wh2, a_dst, fo, nullptr, kb);

        const int segtot = NNODES * fo;
        init_seg<<<(segtot + 255) / 256, 256, 0, stream>>>(mkey, sden, numb, segtot);

        const int etot4 = ETOT * (fo >> 2);
        edge_max4<<<(etot4 + 255) / 256, 256, 0, stream>>>(
            x, ei, a_src, a_dst, pWp, pbp, mkey, fo);
        edge_acc4<<<(etot4 + 255) / 256, 256, 0, stream>>>(
            x, ei, a_src, a_dst, vbuf, pWp, pbp, mkey, sden, numb, fo);

        finalize_layer<<<(segtot + 255) / 256, 256, 0, stream>>>(
            numb, sden, fh, fo, OFF[L]);
    }

    // MLP: concat(400) -> 64 (WMMA, K padded to 416) -> 8 (scalar)
    {
        const float* Wm1 = (const float*)d_in[22];
        const float* bm1 = (const float*)d_in[23];
        const float* Wm2 = (const float*)d_in[24];
        const float* bm2 = (const float*)d_in[25];

        const int wtot = 416 * 64;
        cvt_w_packed<<<(wtot + 255) / 256, 256, 0, stream>>>(Wm1, wh0, 400, 64, 13);
        launch_gemm(wh0, hid, 64, bm1, 13);

        mlp_out<<<(NNODES + 255) / 256, 256, 0, stream>>>(hid, Wm2, bm2, (float*)d_out);
    }
}
